// MoETransformer_5128190951547
// MI455X (gfx1250) — matrix operand order repaired and checked by reading, not compiled, not run on hardware
//
#include <hip/hip_runtime.h>
#include <stdint.h>

// ---------------------------------------------------------------------------
// MoE transformer forward for MI455X (gfx1250, wave32, WMMA).
//   h   = emb[x]                      [T, D]
//   moe = top2-routed FFN(h)          [T, D]   (bf16 WMMA, fp32 accum)
//   out = moe @ Wo + bo               [T, V]   (bf16 WMMA, fp32 accum)
// ---------------------------------------------------------------------------

#define TOK   2048      // B*S
#define DIM   1024      // D
#define NEXP  16        // E
#define FDIM  4096      // F
#define VOC   32000     // V

typedef __attribute__((ext_vector_type(16))) __bf16 v16bf;
typedef __attribute__((ext_vector_type(8)))  __bf16 v8bf;
typedef __attribute__((ext_vector_type(8)))  float  v8f;
typedef __attribute__((ext_vector_type(4)))  float  v4f;

union BF16x16 { v16bf v; v8bf h[2]; uint32_t u[8]; };

static __device__ __forceinline__ uint32_t fbits(float f) {
    union { float f; uint32_t u; } x; x.f = f; return x.u;
}

// fp32 -> bf16 RNE (used only outside GEMM inner loops)
static __device__ __forceinline__ __bf16 f2bf(float f) {
    uint32_t u = fbits(f);
    uint32_t r = u + 0x7FFFu + ((u >> 16) & 1u);
    uint16_t h = (uint16_t)(r >> 16);
    __bf16 b;
    __builtin_memcpy(&b, &h, 2);
    return b;
}

// Pack 16 fp32 down a column (stride ld, K rows 0..7 and 16..23) -> v16bf by truncation.
static __device__ __forceinline__ v16bf pack16(const float* __restrict__ p, int ld) {
    BF16x16 b;
#pragma unroll
    for (int q = 0; q < 8; ++q) {
        const int k = 2 * q + (q >= 4 ? 8 : 0);
        b.u[q] = __builtin_amdgcn_perm(fbits(p[(size_t)(k + 1) * ld]), fbits(p[(size_t)k * ld]), 0x07060302u);
    }
    return b.v;
}

// A fragment per ISA 16-bit 16x32 layout: lane holds row m = lane&15, the two
// 8-element K-halves selected by lane>>4 -> two contiguous 16B LDS loads.
static __device__ __forceinline__ v16bf afrag(const __bf16* __restrict__ row,
                                              int k0, int kb) {
    BF16x16 a;
    a.h[0] = *(const v8bf*)(row + k0 + kb);        // ds_load_b128
    a.h[1] = *(const v8bf*)(row + k0 + 16 + kb);   // ds_load_b128
    return a.v;
}

#define WMMA_BF16(A, B, C) \
    __builtin_amdgcn_wmma_f32_16x16x32_bf16(false, (A), false, (B), (short)0, (C), false, false)

// Async HBM -> LDS 16B copy (VGLOBAL encoding, tracked by ASYNCcnt).
// Low 32 bits of a generic shared-memory address == LDS byte offset.
static __device__ __forceinline__ void async_b128(uint32_t lds_off, const void* gaddr) {
    asm volatile("global_load_async_to_lds_b128 %0, %1, off"
                 :: "v"(lds_off), "v"(gaddr) : "memory");
}
static __device__ __forceinline__ void wait_async0() {
    asm volatile("s_wait_asynccnt 0x0" ::: "memory");
}

// ---------------------------------------------------------------------------
// Kernel 0: zero the per-expert token counters.
// ---------------------------------------------------------------------------
__global__ void k_zero_counts(int* counts) {
    if (threadIdx.x < NEXP) counts[threadIdx.x] = 0;
}

// ---------------------------------------------------------------------------
// Kernel 1: per-token block (256 thr = 8 waves).
// ---------------------------------------------------------------------------
__global__ void k_embed_gate(const long long* __restrict__ x,
                             const float* __restrict__ emb,
                             const float* __restrict__ Wg,
                             __bf16* __restrict__ ht,
                             float*  __restrict__ moe,
                             int*    __restrict__ counts,
                             int*    __restrict__ tokidx,
                             float*  __restrict__ gatew) {
    const int t   = blockIdx.x;
    const int tid = threadIdx.x;
    const long long v = x[t];
    const float* row = emb + (size_t)v * DIM;

    __shared__ float red[256];
    __shared__ float logits[NEXP];

    float acc[NEXP];
#pragma unroll
    for (int e = 0; e < NEXP; ++e) acc[e] = 0.f;

    for (int i = tid; i < DIM; i += 256) {
        const float f = row[i];
        ht [(size_t)t * DIM + i] = f2bf(f);
        moe[(size_t)t * DIM + i] = 0.f;
        const float* wg = Wg + (size_t)i * NEXP;   // Wg is [D, E] row-major
#pragma unroll
        for (int e = 0; e < NEXP; ++e) acc[e] += f * wg[e];
    }

    for (int e = 0; e < NEXP; ++e) {
        red[tid] = acc[e];
        __syncthreads();
        for (int s = 128; s > 0; s >>= 1) {
            if (tid < s) red[tid] += red[tid + s];
            __syncthreads();
        }
        if (tid == 0) logits[e] = red[0];
        __syncthreads();
    }

    if (tid == 0) {
        int i0 = 0;
        for (int e = 1; e < NEXP; ++e) if (logits[e] > logits[i0]) i0 = e;
        int i1 = (i0 == 0) ? 1 : 0;
        for (int e = 0; e < NEXP; ++e)
            if (e != i0 && logits[e] > logits[i1]) i1 = e;
        const float m  = logits[i0];
        const float e1 = __expf(logits[i1] - m);
        const float s  = 1.f + e1;
        const float w0 = 1.f / s, w1 = e1 / s;
        int p0 = atomicAdd(&counts[i0], 1);
        tokidx[i0 * TOK + p0] = t;  gatew[i0 * TOK + p0] = w0;
        int p1 = atomicAdd(&counts[i1], 1);
        tokidx[i1 * TOK + p1] = t;  gatew[i1 * TOK + p1] = w1;
    }
}

// ---------------------------------------------------------------------------
// Kernel 2: routed expert FFN. grid = (tiles=128, E) so that all tile-blocks
// of one expert are dispatch-adjacent and share the expert's 32 MB weight
// panel through the 192 MB L2 (HBM weight traffic ~537 MB instead of ~8 GB).
// Dynamic LDS ~160 KB of the 320 KB WGP LDS. Each wave owns groups of 4
// N-tiles: one A fragment feeds 4 WMMAs.
// ---------------------------------------------------------------------------
__global__ void k_expert_ffn(const __bf16* __restrict__ ht,
                             const float* __restrict__ W1,
                             const float* __restrict__ b1,
                             const float* __restrict__ W2,
                             const float* __restrict__ b2,
                             const int*   __restrict__ counts,
                             const int*   __restrict__ tokidx,
                             const float* __restrict__ gatew,
                             float* __restrict__ moe) {
    extern __shared__ char smem[];
    __bf16* xs   = (__bf16*)smem;                               // 16*DIM
    __bf16* hs   = (__bf16*)(smem + 16 * DIM * 2);              // 16*FDIM
    int*    toks = (int*)  (smem + 16 * DIM * 2 + 16 * FDIM * 2);
    float*  gws  = (float*)(toks + 16);

    const int e    = blockIdx.y;
    const int base = blockIdx.x * 16;
    const int cnt  = counts[e];
    if (base >= cnt) return;

    const int tid  = threadIdx.x;
    const int lane = tid & 31;
    const int wave = tid >> 5;

    if (tid < 16) {
        const int slot = base + tid;
        toks[tid] = (slot < cnt) ? tokidx[e * TOK + slot] : -1;
        gws[tid]  = (slot < cnt) ? gatew [e * TOK + slot] : 0.f;
    }
    __syncthreads();

    // stage the 16xD bf16 activation tile: async HBM->LDS, 16B granules
    {
        const uint32_t xs_off = (uint32_t)(uintptr_t)xs;
        for (int i = tid * 8; i < 16 * DIM; i += 256 * 8) {
            const int r = i >> 10, c = i & (DIM - 1);
            int tk = toks[r]; if (tk < 0) tk = toks[0];   // pad rows: gate==0
            async_b128(xs_off + (uint32_t)i * 2u, ht + (size_t)tk * DIM + c);
        }
        wait_async0();
    }
    __syncthreads();

    const int m_a = lane & 15;             // A-frag row
    const int kb  = (lane >> 4) * 8;       // A-frag K-half select
    const int n_c = lane & 15;             // C/D-frag column
    const int mhi = (lane >> 4) * 8;       // C/D-frag row offset
    const __bf16* xrow = xs + m_a * DIM;
    const __bf16* hrow = hs + m_a * FDIM;

    // ---- GEMM1: [16,1024] @ [1024,4096] -> relu -> hs ----
    const float* W1e = W1 + (size_t)e * DIM * FDIM;
    const float* b1e = b1 + e * FDIM;
    for (int g = wave; g < FDIM / 64; g += 8) {      // 64 groups of 4 tiles
        const int ncol0 = g * 64;
        v8f c[4];
#pragma unroll
        for (int u = 0; u < 4; ++u) {
            const float bias = b1e[ncol0 + u * 16 + n_c];
#pragma unroll
            for (int i = 0; i < 8; ++i) c[u][i] = bias;
        }
        for (int k0 = 0; k0 < DIM; k0 += 32) {
            const v16bf a = afrag(xrow, k0, kb);
            const float* wr = W1e + (size_t)(k0 + kb) * FDIM + ncol0 + n_c;
            c[0] = WMMA_BF16(a, pack16(wr, FDIM),      c[0]);
            c[1] = WMMA_BF16(a, pack16(wr + 16, FDIM), c[1]);
            c[2] = WMMA_BF16(a, pack16(wr + 32, FDIM), c[2]);
            c[3] = WMMA_BF16(a, pack16(wr + 48, FDIM), c[3]);
        }
#pragma unroll
        for (int u = 0; u < 4; ++u) {
#pragma unroll
            for (int i = 0; i < 8; ++i) {
                float h = c[u][i];
                h = h > 0.f ? h : 0.f;               // relu
                hs[(mhi + i) * FDIM + ncol0 + u * 16 + n_c] = f2bf(h);
            }
        }
    }
    __syncthreads();

    // ---- GEMM2: [16,4096] @ [4096,1024] -> scaled scatter-accumulate ----
    const float* W2e = W2 + (size_t)e * FDIM * DIM;
    const float* b2e = b2 + e * DIM;
    for (int g = wave; g < DIM / 64; g += 8) {       // 16 groups of 4 tiles
        const int ncol0 = g * 64;
        v8f c[4];
#pragma unroll
        for (int u = 0; u < 4; ++u) {
            const float bias = b2e[ncol0 + u * 16 + n_c];
#pragma unroll
            for (int i = 0; i < 8; ++i) c[u][i] = bias;
        }
        for (int k0 = 0; k0 < FDIM; k0 += 32) {
            const v16bf a = afrag(hrow, k0, kb);
            const float* wr = W2e + (size_t)(k0 + kb) * DIM + ncol0 + n_c;
            c[0] = WMMA_BF16(a, pack16(wr, DIM),      c[0]);
            c[1] = WMMA_BF16(a, pack16(wr + 16, DIM), c[1]);
            c[2] = WMMA_BF16(a, pack16(wr + 32, DIM), c[2]);
            c[3] = WMMA_BF16(a, pack16(wr + 48, DIM), c[3]);
        }
#pragma unroll
        for (int u = 0; u < 4; ++u) {
#pragma unroll
            for (int i = 0; i < 8; ++i) {
                const int m  = mhi + i;
                const int tk = toks[m];
                if (tk >= 0)
                    atomicAdd(&moe[(size_t)tk * DIM + ncol0 + u * 16 + n_c],
                              gws[m] * c[u][i]);
            }
        }
    }
}

// ---------------------------------------------------------------------------
// Kernel 3: fp32 MoE accumulator -> bf16 for the output GEMM.
// ---------------------------------------------------------------------------
__global__ void k_to_bf16(const float* __restrict__ moe, __bf16* __restrict__ y) {
    const size_t i = (size_t)blockIdx.x * blockDim.x + threadIdx.x;
    if (i < (size_t)TOK * DIM) y[i] = f2bf(moe[i]);
}

// ---------------------------------------------------------------------------
// Kernel 4: output projection [2048,1024] @ [1024,32000] + bo.
// grid = (T/16, V/256): the 128 token-tile blocks sharing one 1 MB Wo column
// panel are dispatch-adjacent -> Wo streams from HBM exactly once (131 MB).
// ---------------------------------------------------------------------------
__global__ void k_outproj(const __bf16* __restrict__ y,
                          const float* __restrict__ Wo,
                          const float* __restrict__ bo,
                          float* __restrict__ out) {
    __shared__ __bf16 ys[16 * DIM];

    const int tid  = threadIdx.x;
    const int lane = tid & 31;
    const int wave = tid >> 5;
    const int mrow  = blockIdx.x * 16;                   // token tile base
    const int ncol0 = blockIdx.y * 256 + wave * 32;      // 2 tiles per wave

    {
        const uint32_t ys_off = (uint32_t)(uintptr_t)ys;
        for (int i = tid * 8; i < 16 * DIM; i += 256 * 8)
            async_b128(ys_off + (uint32_t)i * 2u, y + (size_t)mrow * DIM + i);
        wait_async0();
    }
    __syncthreads();

    const int m_a = lane & 15;
    const int kb  = (lane >> 4) * 8;
    const int n_c = lane & 15;
    const int mhi = (lane >> 4) * 8;
    const __bf16* yrow = ys + m_a * DIM;

    v8f c[2];
#pragma unroll
    for (int u = 0; u < 2; ++u) {
        const float bias = bo[ncol0 + u * 16 + n_c];
#pragma unroll
        for (int i = 0; i < 8; ++i) c[u][i] = bias;
    }

    for (int k0 = 0; k0 < DIM; k0 += 32) {
        const v16bf a = afrag(yrow, k0, kb);
        const float* wr = Wo + (size_t)(k0 + kb) * VOC + ncol0 + n_c;
        c[0] = WMMA_BF16(a, pack16(wr, VOC),      c[0]);
        c[1] = WMMA_BF16(a, pack16(wr + 16, VOC), c[1]);
    }
#pragma unroll
    for (int u = 0; u < 2; ++u)
#pragma unroll
        for (int i = 0; i < 8; ++i)
            out[(size_t)(mrow + mhi + i) * VOC + ncol0 + u * 16 + n_c] = c[u][i];
}

// ---------------------------------------------------------------------------
// launch
// ---------------------------------------------------------------------------
extern "C" void kernel_launch(void* const* d_in, const int* in_sizes, int n_in,
                              void* d_out, int out_size, void* d_ws, size_t ws_size,
                              hipStream_t stream) {
    (void)in_sizes; (void)n_in; (void)out_size; (void)ws_size;

    const long long* x   = (const long long*)d_in[0];   // [B,S] int64
    const float*     emb = (const float*)d_in[1];       // [V,D]
    const float*     Wg  = (const float*)d_in[2];       // [D,E]
    const float*     W1  = (const float*)d_in[3];       // [E,D,F]
    const float*     b1  = (const float*)d_in[4];       // [E,F]
    const float*     W2  = (const float*)d_in[5];       // [E,F,D]
    const float*     b2  = (const float*)d_in[6];       // [E,D]
    const float*     Wo  = (const float*)d_in[7];       // [D,V]
    const float*     bo  = (const float*)d_in[8];       // [V]
    float*           out = (float*)d_out;               // [B,S,V]

    // workspace layout (~17 MB)
    char* ws = (char*)d_ws;
    __bf16* ht     = (__bf16*)ws;                 ws += (size_t)TOK * DIM * 2;  // 4 MB
    float*  moe    = (float*)ws;                  ws += (size_t)TOK * DIM * 4;  // 8 MB
    __bf16* ybf    = (__bf16*)ws;                 ws += (size_t)TOK * DIM * 2;  // 4 MB
    int*    counts = (int*)ws;                    ws += 256;
    int*    tokbuf = (int*)ws;                    ws += (size_t)NEXP * TOK * 4; // 128 KB
    float*  gatbuf = (float*)ws;                  ws += (size_t)NEXP * TOK * 4; // 128 KB

    k_zero_counts<<<1, 32, 0, stream>>>(counts);

    k_embed_gate<<<TOK, 256, 0, stream>>>(x, emb, Wg, ht, moe,
                                          counts, tokbuf, gatbuf);

    const size_t lds2 = (size_t)16 * DIM * 2 + (size_t)16 * FDIM * 2
                      + 16 * sizeof(int) + 16 * sizeof(float);   // ~160 KB
    k_expert_ffn<<<dim3(TOK / 16, NEXP), 256, lds2, stream>>>(
        ht, W1, b1, W2, b2, counts, tokbuf, gatbuf, moe);

    k_to_bf16<<<(TOK * DIM) / 256, 256, 0, stream>>>(moe, ybf);

    k_outproj<<<dim3(TOK / 16, VOC / 256), 256, 0, stream>>>(ybf, Wo, bo, out);
}
